// GraphAttentionLayer_41652592836717
// MI455X (gfx1250) — matrix operand order repaired and checked by reading, not compiled, not run on hardware
//
#include <hip/hip_runtime.h>
#include <stdint.h>

#define N_NODES 8192
#define IN_FEAT 256
#define OUT_FEAT 128

typedef __bf16        v16bf __attribute__((ext_vector_type(16)));
typedef float         v8f   __attribute__((ext_vector_type(8)));
typedef unsigned int  v4u   __attribute__((ext_vector_type(4)));
typedef int           v8i   __attribute__((ext_vector_type(8)));
typedef int           v4i   __attribute__((ext_vector_type(4)));
typedef int           i4    __attribute__((ext_vector_type(4)));

// -------------------------------------------------------------------------
// Kernel A: h_s = X@W_s (fp32), h_n = X@W_n (fp32) and hnT = bf16(h_n)^T
// -------------------------------------------------------------------------
__global__ __launch_bounds__(256) void gat_proj(
    const float* __restrict__ X, const float* __restrict__ Ws,
    const float* __restrict__ Wn, float* __restrict__ hs,
    float* __restrict__ hn, unsigned short* __restrict__ hnT)
{
  __shared__ float Xs[32][IN_FEAT];              // 32 KB
  __shared__ unsigned short Tn[OUT_FEAT][32];    // 8 KB transpose bounce
  const int tid = threadIdx.x;
  const int i0  = blockIdx.x * 32;

  #pragma unroll
  for (int k = 0; k < 32; ++k) {
    int lin = tid + 256 * k;
    Xs[lin >> 8][lin & 255] = X[i0 * IN_FEAT + lin];
  }
  __syncthreads();

  const int c    = tid & 127;
  const int half = tid >> 7;
  float accs[16], accn[16];
  #pragma unroll
  for (int r = 0; r < 16; ++r) { accs[r] = 0.0f; accn[r] = 0.0f; }

  for (int k = 0; k < IN_FEAT; k += 4) {
    float w0 = Ws[(k + 0) * OUT_FEAT + c], w1 = Ws[(k + 1) * OUT_FEAT + c];
    float w2 = Ws[(k + 2) * OUT_FEAT + c], w3 = Ws[(k + 3) * OUT_FEAT + c];
    float u0 = Wn[(k + 0) * OUT_FEAT + c], u1 = Wn[(k + 1) * OUT_FEAT + c];
    float u2 = Wn[(k + 2) * OUT_FEAT + c], u3 = Wn[(k + 3) * OUT_FEAT + c];
    #pragma unroll
    for (int r = 0; r < 16; ++r) {
      const float4 x = *(const float4*)&Xs[half * 16 + r][k];
      accs[r] = fmaf(x.x, w0, fmaf(x.y, w1, fmaf(x.z, w2, fmaf(x.w, w3, accs[r]))));
      accn[r] = fmaf(x.x, u0, fmaf(x.y, u1, fmaf(x.z, u2, fmaf(x.w, u3, accn[r]))));
    }
  }

  #pragma unroll
  for (int r = 0; r < 16; ++r) {
    const int i = i0 + half * 16 + r;
    hs[(long long)i * OUT_FEAT + c] = accs[r];
    hn[(long long)i * OUT_FEAT + c] = accn[r];
    Tn[c][half * 16 + r] = __builtin_bit_cast(unsigned short, (__bf16)accn[r]);
  }
  __syncthreads();
  // Coalesced store of the bf16 transpose tile: hnT[c][i0 .. i0+31]
  #pragma unroll
  for (int k = 0; k < 8; ++k) {
    int lin  = tid + 256 * k;          // 2048 uints total
    int cc   = lin >> 4;
    int part = lin & 15;
    unsigned int val = ((const unsigned int*)&Tn[cc][0])[part];
    ((unsigned int*)(hnT + (long long)cc * N_NODES + i0))[part] = val;
  }
}

// -------------------------------------------------------------------------
// Kernel B: s_i = h_s[i]·a_s, n_i = h_n[i]·a_n, per-block max(n) partials.
// -------------------------------------------------------------------------
__device__ __forceinline__ float dot4(float4 a, float4 b) {
  return fmaf(a.x, b.x, fmaf(a.y, b.y, fmaf(a.z, b.z, a.w * b.w)));
}

__global__ __launch_bounds__(256) void gat_vec(
    const float* __restrict__ hs, const float* __restrict__ hn,
    const float* __restrict__ a_s, const float* __restrict__ a_n,
    float* __restrict__ svec, float* __restrict__ nvec, float* __restrict__ pmax)
{
  const int lane = threadIdx.x & 31;
  const int wid  = threadIdx.x >> 5;
  const float4 asv = *(const float4*)(a_s + lane * 4);
  const float4 anv = *(const float4*)(a_n + lane * 4);
  float nmax = -1e30f;
  for (int it = 0; it < 16; ++it) {
    const int i = blockIdx.x * 128 + it * 8 + wid;
    const float4 h1 = *(const float4*)(hs + (long long)i * OUT_FEAT + lane * 4);
    const float4 h2 = *(const float4*)(hn + (long long)i * OUT_FEAT + lane * 4);
    float sv = dot4(h1, asv);
    float nv = dot4(h2, anv);
    #pragma unroll
    for (int o = 16; o; o >>= 1) {
      sv += __shfl_down(sv, o, 32);
      nv += __shfl_down(nv, o, 32);
    }
    if (lane == 0) {
      svec[i] = sv; nvec[i] = nv;
      nmax = fmaxf(nmax, nv);
    }
  }
  __shared__ float red[8];
  if (lane == 0) red[wid] = nmax;
  __syncthreads();
  if (threadIdx.x == 0) {
    float m = red[0];
    #pragma unroll
    for (int q = 1; q < 8; ++q) m = fmaxf(m, red[q]);
    pmax[blockIdx.x] = m;
  }
}

__global__ __launch_bounds__(64) void gat_max(const float* __restrict__ pmax,
                                              float* __restrict__ maxn)
{
  __shared__ float r[64];
  r[threadIdx.x] = pmax[threadIdx.x];
  __syncthreads();
  if (threadIdx.x == 0) {
    float m = r[0];
    for (int q = 1; q < 64; ++q) m = fmaxf(m, r[q]);
    maxn[0] = m;
  }
}

// -------------------------------------------------------------------------
// Kernel C: fused masked-softmax + attention@h_n + elu, single pass over adj.
// 2 waves/block, 16 rows/wave, K-chunks of 32 via v_wmma_f32_16x16x32_bf16.
// B tiles staged into LDS by the Tensor Data Mover (double buffered,
// s_wait_tensorcnt). All 8 B fragments are loaded before the WMMA burst so
// ds latency pipelines instead of serializing (distinct register blocks).
// Row sums Z = P@ones via a 9th WMMA against a constant all-ones fragment.
// -------------------------------------------------------------------------
#define BROW 40  // LDS B-tile row stride in bf16 units (64B data + 16B TDM pad)

__device__ __forceinline__ void tdm_load_tile(unsigned lds_addr,
                                              const unsigned short* gptr)
{
  const unsigned long long ga = (unsigned long long)(uintptr_t)gptr;
  v4u g0;
  g0[0] = 1u;                                   // count=1 valid descriptor
  g0[1] = lds_addr;                             // LDS byte address
  g0[2] = (unsigned)ga;                         // global_addr[31:0]
  g0[3] = (unsigned)((ga >> 32) & 0x01FFFFFFu)  // global_addr[56:32]
        | (2u << 30);                           // type = 2 (image)
  v8i g1;
  g1[0] = (int)((1u << 16)    // data_size = 1 (2 bytes)
        |       (1u << 20)    // pad_enable
        |       (3u << 22)    // pad_interval code 3 = every 16 dwords
        |       (3u << 25));  // pad_amount  code 3 = 4 dwords
  g1[1] = (int)(8192u << 16); // tensor_dim0 = 8192 (elements)
  g1[2] = (int)(128u  << 16); // tensor_dim1 = 128
  g1[3] = (int)(32u   << 16); // tile_dim0 = 32 (j, 64B rows)
  g1[4] = 128;                // tile_dim1 = 128 (c), tile_dim2 = 0
  g1[5] = 8192;               // tensor_dim0_stride = 8192 elements
  g1[6] = 0;
  g1[7] = 0;
  const v4i z4 = {0, 0, 0, 0};
  const v8i z8 = {0, 0, 0, 0, 0, 0, 0, 0};
  __builtin_amdgcn_tensor_load_to_lds(g0, g1, z4, z4, z8, 0);
}

__global__ __launch_bounds__(64) void gat_attn(
    const int* __restrict__ adj, const float* __restrict__ hs,
    const float* __restrict__ svec, const float* __restrict__ nvec,
    const float* __restrict__ maxn_p, const unsigned short* __restrict__ hnT,
    float* __restrict__ out)
{
  __shared__ __attribute__((aligned(16))) unsigned short ldsB[2][OUT_FEAT * BROW];
  const int lane = threadIdx.x & 31;
  const int wid  = threadIdx.x >> 5;
  const int g    = lane >> 4;     // K half (A/B fragment lane group)
  const int nl   = lane & 15;     // row-within-tile / output column
  const int r0   = blockIdx.x * 32 + wid * 16;
  const int row  = r0 + nl;

  const float maxn = maxn_p[0];
  const float s_i  = svec[row];
  const float m_i  = fmaxf(s_i + maxn, 0.0f);  // relu-monotone softmax bound

  v8f acc[8];
  v8f accz;                        // Z = P @ ones (row sums via matrix unit)
  #pragma unroll
  for (int t = 0; t < 8; ++t)
    #pragma unroll
    for (int k = 0; k < 8; ++k) acc[t][k] = 0.0f;
  #pragma unroll
  for (int k = 0; k < 8; ++k) accz[k] = 0.0f;

  v16bf Bones;                     // constant all-ones B fragment
  #pragma unroll
  for (int e = 0; e < 16; ++e) Bones[e] = (__bf16)1.0f;

  const unsigned lds0 = (unsigned)(uintptr_t)&ldsB[0][0];
  const unsigned lds1 = (unsigned)(uintptr_t)&ldsB[1][0];

  if (wid == 0) tdm_load_tile(lds0, hnT);   // prologue: chunk 0 -> buf 0

  const int* __restrict__ adjRow = adj + (long long)row * N_NODES;
  const int NCH = N_NODES / 32;
  for (int ch = 0; ch < NCH; ++ch) {
    if (wid == 0) __builtin_amdgcn_s_wait_tensorcnt(0);
    __syncthreads();                                   // buf[ch&1] ready
    if (wid == 0 && (ch + 1) < NCH)                    // prefetch next tile
      tdm_load_tile(((ch + 1) & 1) ? lds1 : lds0, hnT + (ch + 1) * 32);

    const int base0 = ch * 32 + 8 * g;
    // adj: one full 128B line per row per chunk, streamed non-temporally
    i4 a0 = __builtin_nontemporal_load((const i4*)(adjRow + base0));
    i4 a1 = __builtin_nontemporal_load((const i4*)(adjRow + base0 + 4));
    i4 b0 = __builtin_nontemporal_load((const i4*)(adjRow + base0 + 16));
    i4 b1 = __builtin_nontemporal_load((const i4*)(adjRow + base0 + 20));
    const float4 n0 = *(const float4*)(nvec + base0);
    const float4 n1 = *(const float4*)(nvec + base0 + 4);
    const float4 n2 = *(const float4*)(nvec + base0 + 16);
    const float4 n3 = *(const float4*)(nvec + base0 + 20);

    // Load ALL 8 B fragments first (distinct registers -> pipelined ds waits)
    union BF { uint4 u[2]; v16bf v; } Bf[8];
    const unsigned short* bufp = ldsB[ch & 1];
    #pragma unroll
    for (int tl = 0; tl < 8; ++tl) {
      const unsigned short* bp = bufp + (16 * tl + nl) * BROW + 8 * g;
      Bf[tl].u[0] = *(const uint4*)(bp);   // K = e + 8g + 8*(e>=8) : two 8-bf16 runs
      Bf[tl].u[1] = *(const uint4*)(bp + 16);
    }

    v16bf A;
    auto pv = [&](int av, float nj) -> __bf16 {
      float t = fmaxf(s_i + nj, 0.0f);              // relu logit
      float p = (av > 0) ? __expf(t - m_i) : 0.0f;  // masked exp (<=1)
      return (__bf16)p;
    };
    // Pack directly in A-fragment element order (K = e + 8g + 8*(e>=8))
    A[0]  = pv(a0[0], n0.x); A[1]  = pv(a0[1], n0.y);
    A[2]  = pv(a0[2], n0.z); A[3]  = pv(a0[3], n0.w);
    A[4]  = pv(a1[0], n1.x); A[5]  = pv(a1[1], n1.y);
    A[6]  = pv(a1[2], n1.z); A[7]  = pv(a1[3], n1.w);
    A[8]  = pv(b0[0], n2.x); A[9]  = pv(b0[1], n2.y);
    A[10] = pv(b0[2], n2.z); A[11] = pv(b0[3], n2.w);
    A[12] = pv(b1[0], n3.x); A[13] = pv(b1[1], n3.y);
    A[14] = pv(b1[2], n3.z); A[15] = pv(b1[3], n3.w);

    // Row sums via the matrix unit: accz += A @ ones
    accz = __builtin_amdgcn_wmma_f32_16x16x32_bf16(
        false, A, false, Bones, (short)0, accz, false, false);

    #pragma unroll
    for (int tl = 0; tl < 8; ++tl) {
      acc[tl] = __builtin_amdgcn_wmma_f32_16x16x32_bf16(
          false, A, false, Bf[tl].v, (short)0, acc[tl], false, false);
    }
  }

  // C/D layout: VGPR v, lane L -> row r0 + v + 8g, col 16*tl + nl.
  // accz[v] (any lane) == Z[r0 + v + 8g]: exactly the Z each lane needs.
  #pragma unroll
  for (int v = 0; v < 8; ++v) {
    const int i = r0 + v + 8 * g;
    const float invZ = 1.0f / fmaxf(accz[v], 1e-30f);
    #pragma unroll
    for (int tl = 0; tl < 8; ++tl) {
      const int cidx = 16 * tl + nl;
      const float hsv = hs[(long long)i * OUT_FEAT + cidx];
      float val = fmaf(acc[tl][v], invZ, 0.5f * hsv);  // attention_s == 0.5
      out[(long long)i * OUT_FEAT + cidx] =
          (val > 0.0f) ? val : (__expf(val) - 1.0f);   // elu
    }
  }
}

// -------------------------------------------------------------------------
extern "C" void kernel_launch(void* const* d_in, const int* in_sizes, int n_in,
                              void* d_out, int out_size, void* d_ws, size_t ws_size,
                              hipStream_t stream)
{
  (void)in_sizes; (void)n_in; (void)out_size; (void)ws_size;
  const float* X   = (const float*)d_in[0];
  const int*   adj = (const int*)d_in[1];
  const float* Ws  = (const float*)d_in[2];
  const float* a_s = (const float*)d_in[3];
  const float* Wn  = (const float*)d_in[4];
  const float* a_n = (const float*)d_in[5];
  float* out = (float*)d_out;

  char* ws = (char*)d_ws;                                  // ~10.1 MB used
  float* hs            = (float*)(ws);                     // 4 MB fp32
  float* hn            = (float*)(ws + (4u << 20));        // 4 MB fp32
  unsigned short* hnT  = (unsigned short*)(ws + (8u << 20)); // 2 MB bf16 (transposed)
  float* svec          = (float*)(ws + (10u << 20));
  float* nvec          = (float*)(ws + (10u << 20) + (32u << 10));
  float* pmax          = (float*)(ws + (10u << 20) + (64u << 10));
  float* maxn          = (float*)(ws + (10u << 20) + (64u << 10) + 4096);

  hipLaunchKernelGGL(gat_proj, dim3(N_NODES / 32), dim3(256), 0, stream,
                     X, Ws, Wn, hs, hn, hnT);
  hipLaunchKernelGGL(gat_vec, dim3(64), dim3(256), 0, stream,
                     hs, hn, a_s, a_n, svec, nvec, pmax);
  hipLaunchKernelGGL(gat_max, dim3(1), dim3(64), 0, stream, pmax, maxn);
  hipLaunchKernelGGL(gat_attn, dim3(N_NODES / 32), dim3(64), 0, stream,
                     adj, hs, svec, nvec, maxn, hnT, out);
}
